// DynamicRadiusChannelFusion_33208687133304
// MI455X (gfx1250) — matrix operand order repaired and checked by reading, not compiled, not run on hardware
//
#include <hip/hip_runtime.h>
#include <math.h>

typedef __attribute__((ext_vector_type(16))) _Float16 v16h;
typedef __attribute__((ext_vector_type(8)))  float    v8f;

constexpr int kB = 4;
constexpr int kN = 16384;
constexpr int kM = 4096;
constexpr int kC = 128;
constexpr int kK = 16;           // neighbors
constexpr float kMaxR = 0.3f;

// ---------------------------------------------------------------------------
// Kernel 0: convert W1 (256x128) and W2 (128x128) to f16 in CDNA5 WMMA
// B-matrix layout in this code's contiguous k order (A uses the same): flat = ((q*8 + t)*32 + lane)*16 + h
//   element = W[(32q + h + 16*(lane/16)) * 128 + (16t + lane%16)]
// so each wave lane later loads a contiguous 32-byte v16h fragment.
// ---------------------------------------------------------------------------
__global__ void swizzle_weights(const float* __restrict__ W1,
                                const float* __restrict__ W2,
                                _Float16* __restrict__ w1h,
                                _Float16* __restrict__ w2h) {
  int id = blockIdx.x * blockDim.x + threadIdx.x;
  if (id < 32768) {                       // W1h: q=0..7, t=0..7
    int h = id & 15, l = (id >> 4) & 31, t = (id >> 9) & 7, q = id >> 12;
    int row = 32 * q + h + 16 * (l >> 4);
    int col = 16 * t + (l & 15);
    w1h[id] = (_Float16)W1[row * kC + col];
  } else if (id < 49152) {                // W2h: q=0..3, t=0..7
    int f = id - 32768;
    int h = f & 15, l = (f >> 4) & 31, t = (f >> 9) & 7, q = f >> 12;
    int row = 32 * q + h + 16 * (l >> 4);
    int col = 16 * t + (l & 15);
    w2h[f] = (_Float16)W2[row * kC + col];
  }
}

// ---------------------------------------------------------------------------
// Kernel 1: radius-kNN, one wave32 per center.
// Key = (float_bits(masked_dist) << 32) | point_index  (monotonic, exact
// top_k tie-break: smaller distance first, then smaller index).
// ---------------------------------------------------------------------------
__global__ void __launch_bounds__(256) knn_kernel(const float* __restrict__ points,
                                                  const int* __restrict__ center_idx,
                                                  int* __restrict__ knn_out) {
  __shared__ unsigned long long mergeBuf[8 * 512];   // 32 KB
  const int lane = threadIdx.x & 31;
  const int wave = threadIdx.x >> 5;
  const int center = blockIdx.x * 8 + wave;
  const int b = center / kM;
  const int m = center - b * kM;
  const int ci = center_idx[b * kM + m];
  const float* pb = points + (size_t)b * kN * 3;
  const float cx = pb[ci * 3 + 0], cy = pb[ci * 3 + 1], cz = pb[ci * 3 + 2];
  const float a_sq = cx * cx + cy * cy + cz * cz;

  unsigned long long keys[16];
#pragma unroll
  for (int i = 0; i < 16; ++i) keys[i] = ~0ull;
  unsigned long long curMax = ~0ull;
  int argMax = 0;

  for (int it = 0; it < kN / 32; ++it) {
    int n = it * 32 + lane;
    float px = pb[n * 3 + 0], py = pb[n * 3 + 1], pz = pb[n * 3 + 2];
    float b_sq = px * px + py * py + pz * pz;
    float inner = cx * px + cy * py + cz * pz;
    float d2 = fmaxf(a_sq + b_sq - 2.0f * inner, 0.0f);
    float dist = sqrtf(d2 + 1e-8f);
    float masked = (dist <= kMaxR) ? dist : 1e9f;
    unsigned long long key =
        ((unsigned long long)__float_as_uint(masked) << 32) | (unsigned)n;
    if (key < curMax) {
      keys[argMax] = key;
      curMax = 0; argMax = 0;
#pragma unroll
      for (int i = 0; i < 16; ++i)
        if (keys[i] > curMax) { curMax = keys[i]; argMax = i; }
    }
  }

  unsigned long long* buf = mergeBuf + wave * 512;
#pragma unroll
  for (int i = 0; i < 16; ++i) buf[lane * 16 + i] = keys[i];
  __syncthreads();

  int myIdx = 0;
  for (int r = 0; r < 16; ++r) {
    unsigned long long lmin = ~0ull; int lslot = 0;
#pragma unroll
    for (int j = 0; j < 16; ++j) {
      unsigned long long v = buf[j * 32 + lane];
      if (v < lmin) { lmin = v; lslot = j * 32 + lane; }
    }
    unsigned long long gmin = lmin;
#pragma unroll
    for (int msk = 16; msk >= 1; msk >>= 1) {
      unsigned long long o = __shfl_xor(gmin, msk, 32);
      gmin = (o < gmin) ? o : gmin;
    }
    if (lmin == gmin) buf[lslot] = ~0ull;   // unique keys -> single owner
    if (lane == r) myIdx = (int)(unsigned)(gmin & 0xffffffffu);
    __syncthreads();
  }
  if (lane < 16) knn_out[center * kK + lane] = myIdx;
}

// ---------------------------------------------------------------------------
// Kernel 2: fused gather + LayerNorm + (16x256)@(256x128) WMMA + ReLU +
// (16x128)@(128x128) WMMA + sigmoid gate + K-mean + residual + GEMV + ReLU.
// One workgroup (8 waves) per center; wave w owns output-column tile w.
// ---------------------------------------------------------------------------
__global__ void __launch_bounds__(256) fuse_kernel(
    const float* __restrict__ feats, const int* __restrict__ center_idx,
    const float* __restrict__ ln_g, const float* __restrict__ ln_b,
    const _Float16* __restrict__ w1h, const float* __restrict__ b1,
    const _Float16* __restrict__ w2h, const float* __restrict__ b2,
    const float* __restrict__ W3, const float* __restrict__ b3,
    const int* __restrict__ knn_idx, float* __restrict__ out) {
  __shared__ float cf[kC];
  __shared__ float neighF[kK][kC];
  __shared__ __align__(32) _Float16 swA[8 * 32 * 16];  // GEMM1 A, swizzled
  __shared__ __align__(32) _Float16 hA[4 * 32 * 16];   // GEMM2 A, swizzled
  __shared__ float fusedV[kC];
  __shared__ int kidx[kK];

  const int tid = threadIdx.x;
  const int lane = tid & 31;
  const int wave = tid >> 5;
  const int center = blockIdx.x;
  const int b = center / kM;
  const int ci = center_idx[center];

  if (tid < kC) cf[tid] = feats[((size_t)b * kN + ci) * kC + tid];
  if (tid < kK) kidx[tid] = knn_idx[center * kK + tid];
  __syncthreads();

  const int row = tid >> 4;        // 0..15 : combo row (neighbor k)
  const int c0 = tid & 15;
  {
    const float* fr = feats + ((size_t)b * kN + kidx[row]) * kC;
#pragma unroll
    for (int j = 0; j < 8; ++j) neighF[row][c0 + 16 * j] = fr[c0 + 16 * j];
  }
  __syncthreads();

  // ---- LayerNorm over 256 (two-pass), write normed f16 into A-layout ----
  float x[16];
#pragma unroll
  for (int j = 0; j < 16; ++j) {
    int col = c0 + 16 * j;
    x[j] = (col < kC) ? cf[col] : neighF[row][col - kC];
  }
  float s = 0.f;
#pragma unroll
  for (int j = 0; j < 16; ++j) s += x[j];
#pragma unroll
  for (int msk = 8; msk >= 1; msk >>= 1) s += __shfl_xor(s, msk, 32);
  float mu = s * (1.0f / 256.0f);
  float v = 0.f;
#pragma unroll
  for (int j = 0; j < 16; ++j) { float d = x[j] - mu; v += d * d; }
#pragma unroll
  for (int msk = 8; msk >= 1; msk >>= 1) v += __shfl_xor(v, msk, 32);
  float rstd = rsqrtf(v * (1.0f / 256.0f) + 1e-5f);
#pragma unroll
  for (int j = 0; j < 16; ++j) {
    int col = c0 + 16 * j;
    float nv = (x[j] - mu) * rstd * ln_g[col] + ln_b[col];
    int q = col >> 5, kl = col & 31;
    int g = kl >> 4, hh = kl & 15;
    swA[(q * 32 + g * 16 + row) * 16 + hh] = (_Float16)nv;
  }
  __syncthreads();

  // ---- GEMM1: normed(16x256) @ W1(256x128), wave owns N-tile t=wave ----
  const int t = wave;
  v8f acc;
#pragma unroll
  for (int i = 0; i < 8; ++i) acc[i] = 0.0f;
#pragma unroll
  for (int q = 0; q < 8; ++q) {
    v16h a = *(const v16h*)(&swA[(q * 32 + lane) * 16]);
    v16h bb = *(const v16h*)(w1h + ((q * 8 + t) * 32 + lane) * 16);
    acc = __builtin_amdgcn_wmma_f32_16x16x32_f16(false, a, false, bb,
                                                 (short)0, acc, false, false);
  }
  const int col = 16 * t + (lane & 15);
  const float bias1 = b1[col];
#pragma unroll
  for (int i = 0; i < 8; ++i) {
    float hv = fmaxf(acc[i] + bias1, 0.0f);
    int r2 = i + 8 * (lane >> 4);            // matrix row (neighbor k)
    int q2 = col >> 5, kl = col & 31;
    int g = kl >> 4, hh = kl & 15;
    hA[(q2 * 32 + g * 16 + r2) * 16 + hh] = (_Float16)hv;
  }
  __syncthreads();

  // ---- GEMM2: h(16x128) @ W2(128x128) ----
  v8f acc2;
#pragma unroll
  for (int i = 0; i < 8; ++i) acc2[i] = 0.0f;
#pragma unroll
  for (int q = 0; q < 4; ++q) {
    v16h a = *(const v16h*)(&hA[(q * 32 + lane) * 16]);
    v16h bb = *(const v16h*)(w2h + ((q * 8 + t) * 32 + lane) * 16);
    acc2 = __builtin_amdgcn_wmma_f32_16x16x32_f16(false, a, false, bb,
                                                  (short)0, acc2, false, false);
  }
  const float bias2 = b2[col];
  float wsum = 0.f;
#pragma unroll
  for (int i = 0; i < 8; ++i) {
    int r2 = i + 8 * (lane >> 4);
    float gate = 1.0f / (1.0f + expf(-(acc2[i] + bias2)));
    wsum += gate * neighF[r2][col];
  }
  wsum += __shfl_xor(wsum, 16, 32);          // combine row halves (same col)
  if (lane < 16) fusedV[col] = wsum * (1.0f / 16.0f) + cf[col];
  __syncthreads();

  // ---- GEMV: out = relu(fused(128) @ W3(128x128) + b3) ----
  if (tid < kC) {
    float a3 = b3[tid];
#pragma unroll 8
    for (int e = 0; e < kC; ++e) a3 = fmaf(fusedV[e], W3[e * kC + tid], a3);
    out[(size_t)center * kC + tid] = fmaxf(a3, 0.0f);
  }
}

// ---------------------------------------------------------------------------
extern "C" void kernel_launch(void* const* d_in, const int* in_sizes, int n_in,
                              void* d_out, int out_size, void* d_ws, size_t ws_size,
                              hipStream_t stream) {
  const float* points = (const float*)d_in[0];
  const float* feats  = (const float*)d_in[1];
  const int*   cidx   = (const int*)d_in[2];
  const float* ln_g   = (const float*)d_in[3];
  const float* ln_b   = (const float*)d_in[4];
  const float* W1     = (const float*)d_in[5];
  const float* b1     = (const float*)d_in[6];
  const float* W2     = (const float*)d_in[7];
  const float* b2     = (const float*)d_in[8];
  const float* W3     = (const float*)d_in[9];
  const float* b3     = (const float*)d_in[10];

  float* out = (float*)d_out;
  int* knn = (int*)(out + (size_t)kB * kM * kC);   // second tuple output
  _Float16* w1h = (_Float16*)d_ws;                 // 64 KB
  _Float16* w2h = w1h + 32768;                     // 32 KB

  swizzle_weights<<<192, 256, 0, stream>>>(W1, W2, w1h, w2h);
  knn_kernel<<<(kB * kM) / 8, 256, 0, stream>>>(points, cidx, knn);
  fuse_kernel<<<kB * kM, 256, 0, stream>>>(feats, cidx, ln_g, ln_b,
                                           w1h, b1, w2h, b2, W3, b3, knn, out);
}
